// Attention_50044958933391
// MI455X (gfx1250) — matrix operand order repaired and checked by reading, not compiled, not run on hardware
//
#include <hip/hip_runtime.h>

// ---------------------------------------------------------------------------
// Problem constants (B, N, C, H) = (4, 512, 384, 6), RATE=0.25 -> NSUB=128
// ---------------------------------------------------------------------------
constexpr int Bb   = 4;
constexpr int Nn   = 512;
constexpr int Cc   = 384;
constexpr int Hh   = 6;
constexpr int NSUB = 128;
constexpr int BN   = Bb * Nn;    // 2048 rows
constexpr int C3   = 3 * Cc;     // 1152 qkv cols
constexpr float SCALE = 0.125f;  // (C/H)^-0.5 = 64^-0.5

typedef __attribute__((ext_vector_type(16))) _Float16 v16h;
typedef __attribute__((ext_vector_type(8)))  _Float16 v8h;
typedef __attribute__((ext_vector_type(8)))  float    v8f;
typedef __attribute__((ext_vector_type(2)))  float    v2f;
typedef __attribute__((ext_vector_type(4)))  int      v4i;

// ---------------------------------------------------------------------------
// Async Global->LDS copies (CDNA5, ASYNCcnt) with graceful fallback.
// Builtin signatures are typed: b32 takes global int* / local int*,
// b128 takes global v4i* / local v4i*.
// ---------------------------------------------------------------------------
#if defined(__HIP_DEVICE_COMPILE__) &&                                      \
    __has_builtin(__builtin_amdgcn_global_load_async_to_lds_b32) &&         \
    __has_builtin(__builtin_amdgcn_global_load_async_to_lds_b128) &&        \
    __has_builtin(__builtin_amdgcn_s_wait_asynccnt)
#define HAS_ASYNC_LDS 1
#else
#define HAS_ASYNC_LDS 0
#endif

#define LDS_AS __attribute__((address_space(3)))
#define GLB_AS __attribute__((address_space(1)))

#if HAS_ASYNC_LDS
__device__ __forceinline__ void async_copy_b32(void* lds, const void* g) {
    __builtin_amdgcn_global_load_async_to_lds_b32(
        (GLB_AS int*)(uintptr_t)g,
        (LDS_AS int*)(uint32_t)(uintptr_t)lds, 0, 0);
}
__device__ __forceinline__ void async_copy_b128(void* lds, const void* g) {
    __builtin_amdgcn_global_load_async_to_lds_b128(
        (GLB_AS v4i*)(uintptr_t)g,
        (LDS_AS v4i*)(uint32_t)(uintptr_t)lds, 0, 0);
}
__device__ __forceinline__ void async_wait0() { __builtin_amdgcn_s_wait_asynccnt(0); }
#else
__device__ __forceinline__ void async_copy_b32(void* lds, const void* g) {
    *(float*)lds = *(const float*)g;
}
__device__ __forceinline__ void async_copy_b128(void* lds, const void* g) {
    *(float4*)lds = *(const float4*)g;
}
__device__ __forceinline__ void async_wait0() {}
#endif

// ---------------------------------------------------------------------------
// 1. centers: rel[b, 0, m, 0:3] -> centers[(b*N+m)*3 + d]
// ---------------------------------------------------------------------------
__global__ void centers_kernel(const float* __restrict__ rel, float* __restrict__ centers) {
    int t = blockIdx.x * blockDim.x + threadIdx.x;
    if (t >= Bb * Nn) return;
    int b = t / Nn, m = t % Nn;
    const float* src = rel + (((size_t)b * Nn + 0) * Nn + m) * 12;
    centers[t * 3 + 0] = src[0];
    centers[t * 3 + 1] = src[1];
    centers[t * 3 + 2] = src[2];
}

// ---------------------------------------------------------------------------
// 2. kNN: one block per (b,n). Bitonic sort 512 (dist, idx) pairs in LDS,
//    ties broken by lower index (matches top_k(-dist) ordering). Keep 128.
// ---------------------------------------------------------------------------
__launch_bounds__(256)
__global__ void knn_kernel(const float* __restrict__ centers, int* __restrict__ nn) {
    __shared__ float key[Nn];
    __shared__ int   val[Nn];
    int bn = blockIdx.x;
    int b  = bn / Nn, n = bn % Nn;
    int tid = threadIdx.x;

    float cx = centers[(b * Nn + n) * 3 + 0];
    float cy = centers[(b * Nn + n) * 3 + 1];
    float cz = centers[(b * Nn + n) * 3 + 2];

    for (int m = tid; m < Nn; m += 256) {
        float dx = centers[(b * Nn + m) * 3 + 0] - cx;
        float dy = centers[(b * Nn + m) * 3 + 1] - cy;
        float dz = centers[(b * Nn + m) * 3 + 2] - cz;
        float d2 = dx * dx + dy * dy + dz * dz;
        key[m] = sqrtf(fmaxf(d2, 1e-12f));
        val[m] = m;
    }
    __syncthreads();

    for (int k = 2; k <= Nn; k <<= 1) {
        for (int j = k >> 1; j > 0; j >>= 1) {
            for (int i = tid; i < Nn; i += 256) {
                int ixj = i ^ j;
                if (ixj > i) {
                    bool up = ((i & k) == 0);
                    float a = key[i], c = key[ixj];
                    int ia = val[i], ic = val[ixj];
                    bool gt = (a > c) || (a == c && ia > ic);
                    bool lt = (a < c) || (a == c && ia < ic);
                    if (up ? gt : lt) {
                        key[i] = c; key[ixj] = a;
                        val[i] = ic; val[ixj] = ia;
                    }
                }
            }
            __syncthreads();
        }
    }
    if (tid < NSUB) nn[bn * NSUB + tid] = val[tid];
}

// ---------------------------------------------------------------------------
// 3a. f32 -> f16 convert (row-major copy, used for x)
// ---------------------------------------------------------------------------
__global__ void cvt_f32_f16(const float* __restrict__ src, _Float16* __restrict__ dst, int n) {
    int t = blockIdx.x * blockDim.x + threadIdx.x;
    if (t < n) dst[t] = (_Float16)src[t];
}

// 3b. f32 -> f16 convert + transpose: src (R x Ccols) -> dst (Ccols x R)
//     so GEMM B-fragments become contiguous 32B loads per lane.
__global__ void cvt_transpose_f16(const float* __restrict__ src, _Float16* __restrict__ dst,
                                  int R, int Ccols) {
    int t = blockIdx.x * blockDim.x + threadIdx.x;
    if (t >= R * Ccols) return;
    int r = t / Ccols, c = t % Ccols;
    dst[(size_t)c * R + r] = (_Float16)src[t];
}

// ---------------------------------------------------------------------------
// 4. WMMA f16 GEMM:  C(MxN) = A(MxK,row) * BT(NxK,row == B col-major) [+bias]
//    One wave owns a 16x64 strip: 1 A-fragment reused for 4 B-fragments.
//    A/B use contiguous k = 16*(lane/16) + s. All fragment loads are wide
//    contiguous vector loads (A: 2x16B, B: 1x32B per lane).
// ---------------------------------------------------------------------------
__device__ __forceinline__ v16h load_a_frag(const _Float16* __restrict__ A, int lda,
                                            int mBase, int k0, int lane) {
    int m  = mBase + (lane & 15);
    int kb = k0 + ((lane >> 4) << 4);                    // halfSel * 16
    const _Float16* p = A + (size_t)m * lda + kb;
    v8h lo = *(const v8h*)(p);                           // K = kb .. kb+7
    v8h hi = *(const v8h*)(p + 8);                       // K = kb+8 .. kb+15
    return __builtin_shufflevector(lo, hi, 0, 1, 2, 3, 4, 5, 6, 7,
                                           8, 9, 10, 11, 12, 13, 14, 15);
}

__device__ __forceinline__ v16h load_b_frag(const _Float16* __restrict__ BT, int ldbt,
                                            int k0, int nBase, int lane) {
    int n  = nBase + (lane & 15);
    int kb = k0 + ((lane >> 4) << 4);                    // kHalf * 16
    return *(const v16h*)(BT + (size_t)n * ldbt + kb);   // 16 contiguous K halves
}

template <bool F16OUT>
__launch_bounds__(256)
__global__ void wmma_gemm(const _Float16* __restrict__ A, const _Float16* __restrict__ BT,
                          int M, int N, int K,
                          _Float16* __restrict__ Ch, float* __restrict__ Cf,
                          const float* __restrict__ bias) {
    int gwave = (blockIdx.x * blockDim.x + threadIdx.x) >> 5;
    int lane  = threadIdx.x & 31;
    int strips = N >> 6;                      // 16x64 strips per row-band
    int mTile  = gwave / strips;
    int nStrip = gwave % strips;
    if (mTile >= (M >> 4)) return;            // wave-uniform guard (EXEC stays all-1s)

    int mBase  = mTile << 4;
    int nBase0 = nStrip << 6;

    v8f acc0 = {}, acc1 = {}, acc2 = {}, acc3 = {};
    for (int k0 = 0; k0 < K; k0 += 32) {
        v16h a  = load_a_frag(A, K, mBase, k0, lane);
        v16h b0 = load_b_frag(BT, K, k0, nBase0 +  0, lane);
        v16h b1 = load_b_frag(BT, K, k0, nBase0 + 16, lane);
        v16h b2 = load_b_frag(BT, K, k0, nBase0 + 32, lane);
        v16h b3 = load_b_frag(BT, K, k0, nBase0 + 48, lane);
        acc0 = __builtin_amdgcn_wmma_f32_16x16x32_f16(false, a, false, b0, (short)0, acc0, false, false);
        acc1 = __builtin_amdgcn_wmma_f32_16x16x32_f16(false, a, false, b1, (short)0, acc1, false, false);
        acc2 = __builtin_amdgcn_wmma_f32_16x16x32_f16(false, a, false, b2, (short)0, acc2, false, false);
        acc3 = __builtin_amdgcn_wmma_f32_16x16x32_f16(false, a, false, b3, (short)0, acc3, false, false);
    }

    // D layout: VGPR r -> M = (lane>=16 ? 8 : 0) + r, N = lane & 15
    int nCol = lane & 15;
    int mOff = (lane >> 4) << 3;
    v8f accs[4] = {acc0, acc1, acc2, acc3};
#pragma unroll
    for (int j = 0; j < 4; ++j) {
        int nc = nBase0 + j * 16 + nCol;
#pragma unroll
        for (int r = 0; r < 8; ++r) {
            size_t off = (size_t)(mBase + mOff + r) * N + nc;
            if constexpr (F16OUT) {
                Ch[off] = (_Float16)accs[j][r];
            } else {
                Cf[off] = accs[j][r] + bias[nc];
            }
        }
    }
}

// ---------------------------------------------------------------------------
// 5. Fused attention: one block (256 thr = 8 waves) per query (b,n).
//    Streaming over 8 tiles of 16 neighbors:
//      relQ tile (16x384) via v_wmma_f32_16x16x4_f32 (K=12 = 3 x K4 steps),
//      scores = q.(k+rq)*SCALE via shuffle-reduced dots,
//      online softmax + accumulation of p.(v+rq).
//    W_rel preload and relg gathers use async Global->LDS (ASYNCcnt path).
// ---------------------------------------------------------------------------
__launch_bounds__(256)
__global__ void attn_kernel(const _Float16* __restrict__ qkv, const float* __restrict__ rel,
                            const float* __restrict__ Wrel, const float* __restrict__ brel,
                            const int* __restrict__ nn, _Float16* __restrict__ outh) {
    __shared__ __align__(16) float sWrel[12 * Cc];     // 18 KB
    __shared__ __align__(16) float sBrel[Cc];
    __shared__ float sQ[Cc];
    __shared__ __align__(16) float sRelg[16 * 12];
    __shared__ float sRQ[16 * Cc];       // 24 KB
    __shared__ float sScore[Hh * 16];
    __shared__ float sP[Hh * 16];
    __shared__ int   sIdx[NSUB];
    __shared__ float sAcc[Cc];
    __shared__ float sM[Hh], sD[Hh], sR[Hh];

    int bn  = blockIdx.x;                 // flat (b*N + n) == qkv row of this query
    int b   = bn / Nn;
    int tid = threadIdx.x;
    int lane = tid & 31;
    int wv   = tid >> 5;                  // wave 0..7

    // async preload of W_rel (1152 x b128) and b_rel (96 x b128)
    for (int j = tid; j < (12 * Cc) / 4; j += 256)
        async_copy_b128(&sWrel[j * 4], &Wrel[j * 4]);
    for (int j = tid; j < Cc / 4; j += 256)
        async_copy_b128(&sBrel[j * 4], &brel[j * 4]);
    for (int i = tid; i < Cc; i += 256) {
        sQ[i]   = (float)qkv[(size_t)bn * C3 + i];
        sAcc[i] = 0.0f;
    }
    for (int i = tid; i < NSUB; i += 256) {
        int idx = nn[bn * NSUB + i];
        sIdx[i] = idx;
        // warm L2/L0 for the gathered K-rows (global_prefetch_b8 path)
        __builtin_prefetch(qkv + (size_t)(b * Nn + idx) * C3 + Cc, 0, 1);
    }
    if (tid < Hh) { sM[tid] = -1e30f; sD[tid] = 0.0f; sR[tid] = 1.0f; }
    async_wait0();
    __syncthreads();

    for (int t = 0; t < NSUB / 16; ++t) {
        // --- gather relg tile (16 x 12) via async gather ---
        if (tid < 192) {
            int sl = tid / 12, j = tid % 12;
            int idx = sIdx[t * 16 + sl];
            async_copy_b32(&sRelg[sl * 12 + j], &rel[(((size_t)bn) * Nn + idx) * 12 + j]);
        }
        async_wait0();
        __syncthreads();

        // --- RQ tile = relg(16x12) @ Wrel(12x384) + brel, f32 WMMA K=4 x3 ---
        {
            int m  = lane & 15;
            int ks = (lane >> 4) << 1;           // 0 or 2
#pragma unroll
            for (int jj = 0; jj < 3; ++jj) {     // 8 waves x 3 = 24 N-tiles
                int nb   = (wv * 3 + jj) << 4;
                int ncol = nb + (lane & 15);
                v8f acc = {};
#pragma unroll
                for (int kk = 0; kk < 3; ++kk) {
                    int kbase = kk * 4 + ks;
                    v2f af, bf;
                    af[0] = sRelg[m * 12 + kbase + 0];
                    af[1] = sRelg[m * 12 + kbase + 1];
                    bf[0] = sWrel[(kbase + 0) * Cc + ncol];
                    bf[1] = sWrel[(kbase + 1) * Cc + ncol];
                    acc = __builtin_amdgcn_wmma_f32_16x16x4_f32(false, af, false, bf,
                                                                (short)0, acc, false, false);
                }
                int mo = (lane >> 4) << 3;
#pragma unroll
                for (int r = 0; r < 8; ++r)
                    sRQ[(mo + r) * Cc + ncol] = acc[r] + sBrel[ncol];
            }
        }
        __syncthreads();

        // --- scores: wave wv handles s = 2wv, 2wv+1; lane-strided 384-dot ---
        for (int ss = 0; ss < 2; ++ss) {
            int sl  = wv * 2 + ss;
            int idx = sIdx[t * 16 + sl];
            const _Float16* krow = qkv + (size_t)(b * Nn + idx) * C3 + Cc;
            float hacc[6] = {0.f, 0.f, 0.f, 0.f, 0.f, 0.f};
#pragma unroll
            for (int i = 0; i < 12; ++i) {
                int c = lane + 32 * i;           // head = i >> 1 (64 ch per head)
                float kv = (float)krow[c] + sRQ[sl * Cc + c];
                hacc[i >> 1] += sQ[c] * kv;
            }
#pragma unroll
            for (int h = 0; h < Hh; ++h) {
                float v = hacc[h];
                for (int off = 16; off > 0; off >>= 1) v += __shfl_down(v, off, 32);
                if (lane == 0) sScore[h * 16 + sl] = v * SCALE;
            }
        }
        __syncthreads();

        // --- online softmax state update (one thread per head) ---
        if (tid < Hh) {
            int h = tid;
            float mOld = sM[h], mNew = mOld;
            for (int s = 0; s < 16; ++s) mNew = fmaxf(mNew, sScore[h * 16 + s]);
            float r = __expf(mOld - mNew);
            float dsum = 0.0f;
            for (int s = 0; s < 16; ++s) {
                float p = __expf(sScore[h * 16 + s] - mNew);
                sP[h * 16 + s] = p;
                dsum += p;
            }
            sD[h] = sD[h] * r + dsum;
            sM[h] = mNew;
            sR[h] = r;
        }
        __syncthreads();

        // --- accumulate p . (v + rq): thread-per-channel ---
        for (int c = tid; c < Cc; c += 256) {
            int h = c >> 6;
            float a = sAcc[c] * sR[h];
#pragma unroll
            for (int s = 0; s < 16; ++s) {
                int idx = sIdx[t * 16 + s];
                float vv = (float)qkv[(size_t)(b * Nn + idx) * C3 + 2 * Cc + c] + sRQ[s * Cc + c];
                a += sP[h * 16 + s] * vv;
            }
            sAcc[c] = a;
        }
        __syncthreads();
    }

    for (int c = tid; c < Cc; c += 256) {
        int h = c >> 6;
        outh[(size_t)bn * Cc + c] = (_Float16)(sAcc[c] / sD[h]);
    }
}

// ---------------------------------------------------------------------------
// Host launcher
// ---------------------------------------------------------------------------
extern "C" void kernel_launch(void* const* d_in, const int* in_sizes, int n_in,
                              void* d_out, int out_size, void* d_ws, size_t ws_size,
                              hipStream_t stream) {
    const float* x      = (const float*)d_in[0];
    const float* rel    = (const float*)d_in[1];
    const float* W_qkv  = (const float*)d_in[2];
    const float* W_proj = (const float*)d_in[3];
    const float* b_proj = (const float*)d_in[4];
    const float* W_rel  = (const float*)d_in[5];
    const float* b_rel  = (const float*)d_in[6];
    float* out = (float*)d_out;

    // carve workspace (256B aligned)
    char* p = (char*)d_ws;
    auto carve = [&](size_t bytes) -> void* {
        void* r = (void*)p;
        p += (bytes + 255) & ~(size_t)255;
        return r;
    };
    float*     centers = (float*)    carve((size_t)BN * 3 * 4);
    int*       nnIdx   = (int*)      carve((size_t)BN * NSUB * 4);
    _Float16*  xh      = (_Float16*) carve((size_t)BN * Cc * 2);
    _Float16*  wqkvT   = (_Float16*) carve((size_t)Cc * C3 * 2);   // (3C x C)
    _Float16*  wprojT  = (_Float16*) carve((size_t)Cc * Cc * 2);   // (C x C)
    _Float16*  qkvh    = (_Float16*) carve((size_t)BN * C3 * 2);
    _Float16*  attnh   = (_Float16*) carve((size_t)BN * Cc * 2);

    // 1-2. kNN graph
    centers_kernel<<<(BN + 255) / 256, 256, 0, stream>>>(rel, centers);
    knn_kernel<<<BN, 256, 0, stream>>>(centers, nnIdx);

    // 3. convert (x) and convert+transpose (weights) to f16
    cvt_f32_f16<<<(BN * Cc + 255) / 256, 256, 0, stream>>>(x, xh, BN * Cc);
    cvt_transpose_f16<<<(Cc * C3 + 255) / 256, 256, 0, stream>>>(W_qkv, wqkvT, Cc, C3);
    cvt_transpose_f16<<<(Cc * Cc + 255) / 256, 256, 0, stream>>>(W_proj, wprojT, Cc, Cc);

    // 4. qkv = x @ W_qkv  (2048x384 . 384x1152), f16 out
    {
        int waves = (BN / 16) * (C3 / 64);          // 2304 waves
        wmma_gemm<true><<<waves * 32 / 256, 256, 0, stream>>>(
            xh, wqkvT, BN, C3, Cc, qkvh, nullptr, nullptr);
    }

    // 5. fused relQ + attention
    attn_kernel<<<BN, 256, 0, stream>>>(qkvh, rel, W_rel, b_rel, nnIdx, attnh);

    // 6. out = attn @ W_proj + b_proj  (2048x384 . 384x384), f32 out
    {
        int waves = (BN / 16) * (Cc / 64);          // 768 waves
        wmma_gemm<false><<<waves * 32 / 256, 256, 0, stream>>>(
            attnh, wprojT, BN, Cc, Cc, nullptr, out, b_proj);
    }
}
